// MultiHeadAttentionBase_17102559772825
// MI455X (gfx1250) — compile-verified, matrix operand order checked by reading, not run on hardware
//
#include <hip/hip_runtime.h>
#include <hip/hip_bf16.h>

typedef __attribute__((ext_vector_type(16))) _Float16 v16h;
typedef __attribute__((ext_vector_type(8)))  _Float16 v8h;
typedef __attribute__((ext_vector_type(2)))  _Float16 v2h;
typedef __attribute__((ext_vector_type(8)))  float    v8f;

static constexpr int S  = 2048;
static constexpr int E  = 1024;
static constexpr int H  = 16;
static constexpr int D  = 64;      // head dim
static constexpr int KB = 64;      // keys per block iteration
static constexpr int QT = 256;     // queries per workgroup (8 waves x 32)
static constexpr int PITCH = 72;   // f16 row pitch (144 B = 9*16, keeps 16B alignment)

// softmax runs in exp2 domain: fold log2(e) into the 1/sqrt(64) scale
#define QSCALE   0.18033688f       // 0.125 * log2(e)
#define NEG_BIG  2.0e9f

#define EXP2F(x) __builtin_amdgcn_exp2f(x)   // native v_exp_f32 (exp2)

union F4 { float4 v; float f[4]; };

__device__ __forceinline__ v16h cat8(v8h lo, v8h hi) {
  return __builtin_shufflevector(lo, hi, 0,1,2,3,4,5,6,7,8,9,10,11,12,13,14,15);
}

__device__ __forceinline__ v8f wmma_f16(v16h a, v16h b, v8f c) {
  return __builtin_amdgcn_wmma_f32_16x16x32_f16(false, a, false, b, (short)0, c,
                                                false, false);
}

__global__ __launch_bounds__(256)
void mha_flash_wmma(const float* __restrict__ Qg, const float* __restrict__ Kg,
                    const float* __restrict__ Vg, const int* __restrict__ Mg,
                    float* __restrict__ Og)
{
  // double-buffered tiles
  __shared__ __align__(16) _Float16 Kt[2][KB][PITCH]; // K tile, row-major [key][d]
  __shared__ __align__(16) _Float16 Vt[2][D][PITCH];  // V tile transposed [d][key]
  __shared__ __align__(16) float    MFm[2][KB];       // mask multiplier (0/1)
  __shared__ __align__(16) float    MFb[2][KB];       // mask bias (0 / -2e9)

  const int tid  = threadIdx.x;
  const int lane = tid & 31;
  const int w    = tid >> 5;
  const int n15  = lane & 15;
  const int hf   = lane >> 4;

  const int b = blockIdx.y / H;
  const int h = blockIdx.y % H;
  const int wbase = blockIdx.x * QT + w * 32;   // 32 queries per wave (2 tiles)

  // ---- Q^T B-fragments for both query tiles, scaled into exp2 domain ----
  v16h qf[2][2];
#pragma unroll
  for (int qi = 0; qi < 2; ++qi) {
    const float* qp = Qg + (size_t)(b * S + wbase + 16 * qi + n15) * E + h * D;
#pragma unroll
    for (int dc = 0; dc < 2; ++dc) {
      const int base = 32 * dc + 8 * hf;
      F4 a0, a1, a2, a3;
      a0.v = *(const float4*)(qp + base);
      a1.v = *(const float4*)(qp + base + 4);
      a2.v = *(const float4*)(qp + base + 16);
      a3.v = *(const float4*)(qp + base + 20);
      v16h q;
#pragma unroll
      for (int e = 0; e < 4; ++e) {
        q[e]      = (_Float16)(a0.f[e] * QSCALE);
        q[4 + e]  = (_Float16)(a1.f[e] * QSCALE);
        q[8 + e]  = (_Float16)(a2.f[e] * QSCALE);
        q[12 + e] = (_Float16)(a3.f[e] * QSCALE);
      }
      qf[qi][dc] = q;
    }
  }

  // O^T accumulators: [q-tile][d-block], each 16d x 16q, f32
  v8f ot[2][4];
#pragma unroll
  for (int qi = 0; qi < 2; ++qi)
#pragma unroll
    for (int i = 0; i < 4; ++i) { v8f z = {}; ot[qi][i] = z; }

  float m_run[2] = {-3.0e38f, -3.0e38f};
  float l_run[2] = {0.0f, 0.0f};

  // staging assignments
  const int ks = tid >> 2;             // K: key row 0..63
  const int kd = (tid & 3) * 16;       // K: 16 d-values
  const int vk = (tid & 31) * 2;       // V: key pair
  const int vd = (tid >> 5) * 8;       // V: 8 d-values
  const float* kp0 = Kg + (size_t)(b * S + ks) * E + h * D + kd;
  const float* vp0 = Vg + (size_t)(b * S + vk) * E + h * D + vd;
  const int*   mp  = Mg + b * S;

  constexpr int NBLK = S / KB;

  F4 kr[4], vr[4];
  int mi;

  auto load_tile = [&](int blk) {
    const float* kptr = kp0 + (size_t)blk * KB * E;
    kr[0].v = *(const float4*)(kptr);
    kr[1].v = *(const float4*)(kptr + 4);
    kr[2].v = *(const float4*)(kptr + 8);
    kr[3].v = *(const float4*)(kptr + 12);
    const float* vptr = vp0 + (size_t)blk * KB * E;
    vr[0].v = *(const float4*)(vptr);
    vr[1].v = *(const float4*)(vptr + 4);
    vr[2].v = *(const float4*)(vptr + E);
    vr[3].v = *(const float4*)(vptr + E + 4);
    mi = (tid < KB) ? mp[blk * KB + tid] : 1;
    if (blk + 1 < NBLK) {                 // distance-2 prefetch (global_prefetch_b8)
      __builtin_prefetch(kptr + (size_t)KB * E, 0, 1);
      __builtin_prefetch(vptr + (size_t)KB * E, 0, 1);
    }
  };

  auto store_tile = [&](int buf) {
    v8h h0, h1;
#pragma unroll
    for (int e = 0; e < 4; ++e) {
      h0[e]     = (_Float16)kr[0].f[e];
      h0[4 + e] = (_Float16)kr[1].f[e];
      h1[e]     = (_Float16)kr[2].f[e];
      h1[4 + e] = (_Float16)kr[3].f[e];
    }
    *(v8h*)&Kt[buf][ks][kd]     = h0;
    *(v8h*)&Kt[buf][ks][kd + 8] = h1;
#pragma unroll
    for (int e = 0; e < 4; ++e) {
      v2h p0; p0[0] = (_Float16)vr[0].f[e]; p0[1] = (_Float16)vr[2].f[e];
      v2h p1; p1[0] = (_Float16)vr[1].f[e]; p1[1] = (_Float16)vr[3].f[e];
      *(v2h*)&Vt[buf][vd + e][vk]     = p0;
      *(v2h*)&Vt[buf][vd + 4 + e][vk] = p1;
    }
    if (tid < KB) {
      MFm[buf][tid] = mi ? 1.0f : 0.0f;
      MFb[buf][tid] = mi ? 0.0f : -NEG_BIG;
    }
  };

  load_tile(0);

  for (int blk = 0; blk < NBLK; ++blk) {
    const int buf = blk & 1;
    store_tile(buf);
    __syncthreads();
    if (blk + 1 < NBLK) load_tile(blk + 1);   // completes under the WMMAs below

    // -------- scores S^T = K * Q^T : 4 key tiles x 2 query tiles, K-dim 64 --------
    v8f st[2][4];
#pragma unroll
    for (int qi = 0; qi < 2; ++qi)
#pragma unroll
      for (int t = 0; t < 4; ++t) { v8f z = {}; st[qi][t] = z; }

#pragma unroll
    for (int t = 0; t < 4; ++t) {
#pragma unroll
      for (int dc = 0; dc < 2; ++dc) {
        v8h lo = *(const v8h*)&Kt[buf][16 * t + n15][32 * dc + 8 * hf];
        v8h hi = *(const v8h*)&Kt[buf][16 * t + n15][32 * dc + 16 + 8 * hf];
        v16h af = cat8(lo, hi);                   // reused by both query tiles
        st[0][t] = wmma_f16(af, qf[0][dc], st[0][t]);
        st[1][t] = wmma_f16(af, qf[1][dc], st[1][t]);
      }
    }

    // mask multiplier + bias (precomputed at staging; shared by both query tiles)
    float mfv[4][8], bia[4][8];
#pragma unroll
    for (int t = 0; t < 4; ++t) {
      F4 x, y, u, z;
      x.v = *(const float4*)&MFm[buf][16 * t + 8 * hf];
      y.v = *(const float4*)&MFm[buf][16 * t + 8 * hf + 4];
      u.v = *(const float4*)&MFb[buf][16 * t + 8 * hf];
      z.v = *(const float4*)&MFb[buf][16 * t + 8 * hf + 4];
#pragma unroll
      for (int e = 0; e < 4; ++e) {
        mfv[t][e] = x.f[e]; mfv[t][4 + e] = y.f[e];
        bia[t][e] = u.f[e]; bia[t][4 + e] = z.f[e];
      }
    }

    // ---------------- flash softmax per query tile (exp2 domain, register-only P) --
    v16h pf[2][2];
#pragma unroll
    for (int qi = 0; qi < 2; ++qi) {
      float s[4][8];
#pragma unroll
      for (int t = 0; t < 4; ++t)
#pragma unroll
        for (int j = 0; j < 8; ++j)
          s[t][j] = st[qi][t][j] * mfv[t][j] + bia[t][j];   // single fma per score

      float mx = fmaxf(s[0][0], s[0][1]);
#pragma unroll
      for (int t = 0; t < 4; ++t)
#pragma unroll
        for (int j = (t == 0 ? 2 : 0); j < 8; ++j) mx = fmaxf(mx, s[t][j]);
      mx = fmaxf(mx, __shfl_xor(mx, 16, 32));
      const float m_new = fmaxf(m_run[qi], mx);
      const float alpha = EXP2F(m_run[qi] - m_new);

      float p[4][8];
      float ls = 0.0f;
#pragma unroll
      for (int t = 0; t < 4; ++t)
#pragma unroll
        for (int j = 0; j < 8; ++j) { p[t][j] = EXP2F(s[t][j] - m_new); ls += p[t][j]; }
      ls += __shfl_xor(ls, 16, 32);
      l_run[qi] = l_run[qi] * alpha + ls;
      m_run[qi] = m_new;

#pragma unroll
      for (int dt = 0; dt < 4; ++dt)
#pragma unroll
        for (int j = 0; j < 8; ++j) ot[qi][dt][j] *= alpha;

      // P^T B-fragments: C-layout of S^T tiles == B-layout of P^T chunks
#pragma unroll
      for (int kc = 0; kc < 2; ++kc) {
        v16h pfr;
#pragma unroll
        for (int j = 0; j < 8; ++j) {
          pfr[j]     = (_Float16)p[2 * kc][j];
          pfr[8 + j] = (_Float16)p[2 * kc + 1][j];
        }
        pf[qi][kc] = pfr;
      }
    }

    // -------- O^T += V^T * P^T : 4 d-blocks x 2 K-chunks x 2 query tiles --------
#pragma unroll
    for (int dt = 0; dt < 4; ++dt) {
#pragma unroll
      for (int kc = 0; kc < 2; ++kc) {
        v8h lo = *(const v8h*)&Vt[buf][16 * dt + n15][32 * kc + 8 * hf];
        v8h hi = *(const v8h*)&Vt[buf][16 * dt + n15][32 * kc + 16 + 8 * hf];
        v16h vf = cat8(lo, hi);                   // reused by both query tiles
        ot[0][dt] = wmma_f16(vf, pf[0][kc], ot[0][dt]);
        ot[1][dt] = wmma_f16(vf, pf[1][kc], ot[1][dt]);
      }
    }
  }

  // ---------------- normalize and write context[b, q, h*64 + d] ----------------
#pragma unroll
  for (int qi = 0; qi < 2; ++qi) {
    const float linv = 1.0f / ((l_run[qi] > 0.0f) ? l_run[qi] : 1.0f);
    float* op = Og + (size_t)(b * S + wbase + 16 * qi + n15) * E + h * D + 8 * hf;
#pragma unroll
    for (int dt = 0; dt < 4; ++dt) {
      F4 f0, f1;
#pragma unroll
      for (int e = 0; e < 4; ++e) {
        f0.f[e] = ot[qi][dt][e]     * linv;
        f1.f[e] = ot[qi][dt][4 + e] * linv;
      }
      *(float4*)(op + 16 * dt)     = f0.v;
      *(float4*)(op + 16 * dt + 4) = f1.v;
    }
  }
}

extern "C" void kernel_launch(void* const* d_in, const int* in_sizes, int n_in,
                              void* d_out, int out_size, void* d_ws, size_t ws_size,
                              hipStream_t stream) {
  const float* q    = (const float*)d_in[0];
  const float* k    = (const float*)d_in[1];
  const float* v    = (const float*)d_in[2];
  const int*   mask = (const int*)d_in[3];
  float* out = (float*)d_out;

  const int Btot = in_sizes[0] / (S * E);   // batch (4)
  dim3 grid(S / QT, Btot * H);
  mha_flash_wmma<<<grid, 256, 0, stream>>>(q, k, v, mask, out);
}
